// TransformerUNet_65274912965136
// MI455X (gfx1250) — matrix operand order repaired and checked by reading, not compiled, not run on hardware
//
#include <hip/hip_runtime.h>

// ---------------------------------------------------------------------------
// Types for CDNA5 WMMA (wave32): 16x16x32 bf16 -> f32
// ---------------------------------------------------------------------------
typedef __attribute__((ext_vector_type(16))) __bf16         bf16x16;
typedef __attribute__((ext_vector_type(8)))  float          f32x8;
typedef __attribute__((ext_vector_type(8)))  unsigned short u16x8;

static __device__ __forceinline__ f32x8 f32x8_zero() {
    f32x8 r;
#pragma unroll
    for (int i = 0; i < 8; ++i) r[i] = 0.0f;
    return r;
}

// fp32 -> bf16 bits, round-to-nearest-even
static __device__ __forceinline__ unsigned short f2bf(float f) {
    unsigned u = __float_as_uint(f);
    return (unsigned short)((u + 0x7FFFu + ((u >> 16) & 1u)) >> 16);
}

// Build a 16-element bf16 A/B fragment from two 8-element (16B) LDS chunks.
static __device__ __forceinline__ bf16x16 ld_frag2(const unsigned short* p0,
                                                   const unsigned short* p1) {
    bf16x16 r;
    ((u16x8*)&r)[0] = *(const u16x8*)p0;
    ((u16x8*)&r)[1] = *(const u16x8*)p1;
    return r;
}
static __device__ __forceinline__ bf16x16 ld_frag16(const unsigned short* p) {
    return ld_frag2(p, p + 8);
}

static __device__ __forceinline__ f32x8 wmma_bf16(bf16x16 a, bf16x16 b, f32x8 c) {
    return __builtin_amdgcn_wmma_f32_16x16x32_bf16(
        /*neg_a=*/false, a, /*neg_b=*/false, b,
        /*c_mod=*/(short)0, c, /*reuse_a=*/false, /*reuse_b=*/false);
}

static __device__ __forceinline__ float gelu_exact(float x) {
    return 0.5f * x * (1.0f + erff(x * 0.70710678118654752f));
}

// CDNA5 async global->LDS copy (16B per lane), tracked by ASYNCcnt.
// CDNA5 ISA: GLOBAL_LOAD_ASYNC_TO_LDS_B128, GV mode.
static __device__ __forceinline__ void async_copy_b128(unsigned lds_off,
                                                       const void* gptr) {
    asm volatile("global_load_async_to_lds_b128 %0, %1, off"
                 :: "v"(lds_off), "v"(gptr) : "memory");
}
static __device__ __forceinline__ void wait_asynccnt0() {
    asm volatile("s_wait_asynccnt 0x0" ::: "memory");
}

// ---------------------------------------------------------------------------
// GEMM: C[M,N] = act(A[M,K] @ W[K,N] + bias[N]); C written fp32 or bf16.
// Block: 256 thr (8 waves), tile 128x128; wave -> 32x64 (2x4 WMMA tiles).
// Register double-buffer: global loads for k+32 overlap WMMAs for k.
// A staged [row][k], W staged transposed [n][k]: fragment loads are
// contiguous 16B LDS reads matching the 16-bit WMMA operand layouts.
// ---------------------------------------------------------------------------
__global__ __launch_bounds__(256)
void gemm_bias_kernel(const float* __restrict__ A, const float* __restrict__ W,
                      const float* __restrict__ bias, float* __restrict__ Cf,
                      unsigned short* __restrict__ Cb,
                      int M, int N, int K, int act, int out_bf16) {
    __shared__ __align__(16) unsigned short As[128 * 32];  // 8 KB
    __shared__ __align__(16) unsigned short Ws[128 * 32];  // 8 KB

    const int t     = threadIdx.x;
    const int lane  = t & 31;
    const int wave  = t >> 5;
    const int waveM = wave >> 1;   // 0..3 -> 32 rows each
    const int waveN = wave & 1;    // 0..1 -> 64 cols each
    const int bm    = blockIdx.y * 128;
    const int bn    = blockIdx.x * 128;
    const int lm    = lane & 15;
    const int half  = lane >> 4;

    // staging index maps (constant per thread)
    const int aCol = t & 31,  aRow = t >> 5;    // +8 rows per i
    const int wN   = t & 127, wK   = t >> 7;    // +2 k per i

    f32x8 acc[2][4];
#pragma unroll
    for (int i = 0; i < 2; ++i)
#pragma unroll
        for (int j = 0; j < 4; ++j) acc[i][j] = f32x8_zero();

    // ---- preload k0 = 0 into registers ----
    float a_reg[16], w_reg[16];
#pragma unroll
    for (int i = 0; i < 16; ++i) {
        a_reg[i] = A[(bm + aRow + i * 8) * K + aCol];
        w_reg[i] = W[(wK + i * 2) * N + bn + wN];
    }

    for (int k0 = 0; k0 < K; k0 += 32) {
        // ---- store staged registers to LDS (fp32 -> bf16) ----
#pragma unroll
        for (int i = 0; i < 16; ++i) {
            As[(aRow + i * 8) * 32 + aCol] = f2bf(a_reg[i]);
            Ws[(wN)*32 + wK + i * 2]       = f2bf(w_reg[i]);
        }
        __syncthreads();

        // ---- prefetch k+32 into registers; hint k+64 to L2 ----
        if (k0 + 32 < K) {
#pragma unroll
            for (int i = 0; i < 16; ++i) {
                a_reg[i] = A[(bm + aRow + i * 8) * K + k0 + 32 + aCol];
                w_reg[i] = W[(k0 + 32 + wK + i * 2) * N + bn + wN];
            }
            if (k0 + 64 < K) {
                __builtin_prefetch(&A[(bm + aRow) * K + k0 + 64 + aCol], 0, 3);
                __builtin_prefetch(&W[(k0 + 64 + wK) * N + bn + wN], 0, 3);
            }
        }

        // ---- fragments + 8 WMMAs ----
        bf16x16 af[2], bfg[4];
#pragma unroll
        for (int ti = 0; ti < 2; ++ti) {
            const unsigned short* base = &As[(waveM * 32 + ti * 16 + lm) * 32];
            af[ti] = ld_frag2(base + 8 * half, base + 16 + 8 * half);
        }
#pragma unroll
        for (int tj = 0; tj < 4; ++tj) {
            const unsigned short* base = &Ws[(waveN * 64 + tj * 16 + lm) * 32];
            bfg[tj] = ld_frag2(base + 8 * half, base + 16 + 8 * half);
        }
#pragma unroll
        for (int ti = 0; ti < 2; ++ti)
#pragma unroll
            for (int tj = 0; tj < 4; ++tj)
                acc[ti][tj] = wmma_bf16(af[ti], bfg[tj], acc[ti][tj]);
        __syncthreads();
    }

    // Epilogue: C layout — VGPR r: lanes 0-15 -> M=r, lanes 16-31 -> M=r+8
#pragma unroll
    for (int ti = 0; ti < 2; ++ti)
#pragma unroll
        for (int tj = 0; tj < 4; ++tj) {
            int col = bn + waveN * 64 + tj * 16 + lm;
            float bv = bias[col];
#pragma unroll
            for (int r = 0; r < 8; ++r) {
                int row = bm + waveM * 32 + ti * 16 + r + 8 * half;
                float v = acc[ti][tj][r] + bv;
                if (act) v = gelu_exact(v);
                if (out_bf16) Cb[row * N + col] = f2bf(v);
                else          Cf[row * N + col] = v;
            }
        }
}

// ---------------------------------------------------------------------------
// Flash attention over bf16 Q/K/V: block = (batch, head, 64 q rows), 128 thr.
// Each wave owns 16 q-rows. Streams 64-key tiles; Q/K tiles staged via
// GLOBAL_LOAD_ASYNC_TO_LDS_B128 (ASYNCcnt); V staged transposed manually.
// Online softmax; 16 WMMAs per key tile. Head dim 64, D=1024, scale 1/8.
// ---------------------------------------------------------------------------
__global__ __launch_bounds__(128)
void flash_attn_kernel(const unsigned short* __restrict__ Q,
                       const unsigned short* __restrict__ Kin,
                       const unsigned short* __restrict__ V,
                       const float* __restrict__ mask,
                       float* __restrict__ O, int Lq, int Lk) {
    __shared__ __align__(16) unsigned short Qs[64 * 64];    // [q][d]    8 KB
    __shared__ __align__(16) unsigned short Ks[64 * 64];    // [key][d]  8 KB
    __shared__ __align__(16) unsigned short Vt[64 * 64];    // [d][key]  8 KB
    __shared__ float          Ss[4][16 * 64];               // scores   16 KB
    __shared__ __align__(16) unsigned short Ps[4][16 * 64]; // probs     8 KB
    __shared__ float          rowA[4][16];

    const int t    = threadIdx.x;
    const int lane = t & 31;
    const int w    = t >> 5;        // wave 0..3 -> q rows 16w..16w+15
    const int q0   = blockIdx.x * 64;
    const int h    = blockIdx.y;
    const int b    = blockIdx.z;
    const int lm   = lane & 15;
    const int half = lane >> 4;

    // async-copy lane mapping: 4 rows per wave-issue, 16B chunk per lane
    const int cRow = lane >> 3;            // 0..3
    const int cCh  = (lane & 7) * 8;       // u16 offset of 16B chunk in row
    const unsigned ldsQ = (unsigned)(size_t)(void*)Qs;
    const unsigned ldsK = (unsigned)(size_t)(void*)Ks;

    // ---- stage Q tile (64 x 64) via async DMA ----
#pragma unroll
    for (int it = 0; it < 4; ++it) {
        int r = w * 16 + it * 4 + cRow;    // 0..63
        async_copy_b128(ldsQ + (unsigned)((r * 64 + cCh) * 2),
                        Q + (size_t)(b * Lq + q0 + r) * 1024 + h * 64 + cCh);
    }
    wait_asynccnt0();
    __syncthreads();

    bf16x16 qf[2];
#pragma unroll
    for (int c = 0; c < 2; ++c) {
        const unsigned short* base = &Qs[(w * 16 + lm) * 64 + c * 32];
        qf[c] = ld_frag2(base + 8 * half, base + 16 + 8 * half);
    }

    f32x8 o[4];
#pragma unroll
    for (int i = 0; i < 4; ++i) o[i] = f32x8_zero();
    float mreg = -1e30f, lreg = 0.0f;   // valid in lanes 0..15

    for (int kt = 0; kt < Lk; kt += 64) {
        // ---- stage K tile [key][d] via async DMA ----
#pragma unroll
        for (int it = 0; it < 4; ++it) {
            int r = w * 16 + it * 4 + cRow;
            async_copy_b128(ldsK + (unsigned)((r * 64 + cCh) * 2),
                            Kin + (size_t)(b * Lk + kt + r) * 1024 + h * 64 + cCh);
        }
        // ---- stage V tile transposed [d][key] ----
#pragma unroll
        for (int i = 0; i < 32; ++i) {
            int idx = t + i * 128;
            int d = idx & 63, kr = idx >> 6;
            Vt[d * 64 + kr] = V[(b * Lk + kt + kr) * 1024 + h * 64 + d];
        }
        wait_asynccnt0();
        __syncthreads();

        // ---- S(16x64) = Q(16x64) @ K^T : 4 n-tiles x 2 k-chunks = 8 WMMAs
        f32x8 s[4];
#pragma unroll
        for (int j = 0; j < 4; ++j) s[j] = f32x8_zero();
#pragma unroll
        for (int c = 0; c < 2; ++c)
#pragma unroll
            for (int j = 0; j < 4; ++j) {
                bf16x16 kf = ld_frag2(&Ks[(j * 16 + lm) * 64 + c * 32 + 8 * half], &Ks[(j * 16 + lm) * 64 + c * 32 + 16 + 8 * half]);
                s[j] = wmma_bf16(qf[c], kf, s[j]);
            }

        // ---- scale + mask, spill to LDS for row-wise softmax ----
#pragma unroll
        for (int j = 0; j < 4; ++j) {
            float mval = mask ? mask[b * Lk + kt + j * 16 + lm] : 0.0f;
#pragma unroll
            for (int r = 0; r < 8; ++r)
                Ss[w][(r + 8 * half) * 64 + j * 16 + lm] = s[j][r] * 0.125f + mval;
        }
        __syncthreads();

        // ---- online softmax: lane L<16 owns row L of this wave ----
        if (lane < 16) {
            const int row = lane;
            float tm = -1e30f;
#pragma unroll
            for (int k = 0; k < 64; ++k) tm = fmaxf(tm, Ss[w][row * 64 + k]);
            float nm    = fmaxf(mreg, tm);
            float alpha = __expf(mreg - nm);
            float sum   = 0.0f;
#pragma unroll
            for (int k = 0; k < 64; ++k) {
                float p = __expf(Ss[w][row * 64 + k] - nm);
                Ps[w][row * 64 + k] = f2bf(p);
                sum += p;
            }
            lreg = lreg * alpha + sum;
            mreg = nm;
            rowA[w][row] = alpha;
        }
        __syncthreads();

        // ---- rescale accumulators; O(16x64) += P(16x64) @ V(64x64) ----
        float al[8];
#pragma unroll
        for (int r = 0; r < 8; ++r) al[r] = rowA[w][r + 8 * half];
#pragma unroll
        for (int tt = 0; tt < 4; ++tt)
#pragma unroll
            for (int r = 0; r < 8; ++r) o[tt][r] *= al[r];

#pragma unroll
        for (int c2 = 0; c2 < 2; ++c2) {
            bf16x16 pf = ld_frag2(&Ps[w][lm * 64 + c2 * 32 + 8 * half],
                                  &Ps[w][lm * 64 + c2 * 32 + 16 + 8 * half]);
#pragma unroll
            for (int tt = 0; tt < 4; ++tt) {
                bf16x16 vf = ld_frag2(&Vt[(tt * 16 + lm) * 64 + c2 * 32 + 8 * half], &Vt[(tt * 16 + lm) * 64 + c2 * 32 + 16 + 8 * half]);
                o[tt] = wmma_bf16(pf, vf, o[tt]);
            }
        }
        __syncthreads();
    }

    if (lane < 16) rowA[w][lane] = 1.0f / lreg;   // reuse as 1/l
    __syncthreads();

#pragma unroll
    for (int tt = 0; tt < 4; ++tt)
#pragma unroll
        for (int r = 0; r < 8; ++r) {
            int row = r + 8 * half;
            float val = o[tt][r] * rowA[w][row];
            O[(b * Lq + q0 + w * 16 + row) * 1024 + h * 64 + tt * 16 + lm] = val;
        }
}

// ---------------------------------------------------------------------------
// out = LN(h + res) * g + be  (bias already applied inside GEMM)
// One block (256 thr) per row of 1024; wave32 __shfl_xor reductions.
// ---------------------------------------------------------------------------
static __device__ __forceinline__ float wave_reduce_sum(float v) {
#pragma unroll
    for (int off = 16; off > 0; off >>= 1) v += __shfl_xor(v, off, 32);
    return v;
}

__global__ __launch_bounds__(256)
void ln_residual_kernel(const float* __restrict__ hbuf, const float* __restrict__ res,
                        const float* __restrict__ g, const float* __restrict__ be,
                        float* __restrict__ out) {
    __shared__ float redS[8], redQ[8];
    const int row  = blockIdx.x;
    const int t    = threadIdx.x;
    const int lane = t & 31, w = t >> 5;

    float v[4];
    float s = 0.0f, sq = 0.0f;
#pragma unroll
    for (int i = 0; i < 4; ++i) {
        int col = t + i * 256;
        v[i] = hbuf[row * 1024 + col] + res[row * 1024 + col];
        s  += v[i];
        sq += v[i] * v[i];
    }
    s  = wave_reduce_sum(s);
    sq = wave_reduce_sum(sq);
    if (lane == 0) { redS[w] = s; redQ[w] = sq; }
    __syncthreads();
    float ts = 0.0f, tq = 0.0f;
#pragma unroll
    for (int i = 0; i < 8; ++i) { ts += redS[i]; tq += redQ[i]; }
    const float mu   = ts * (1.0f / 1024.0f);
    const float var  = tq * (1.0f / 1024.0f) - mu * mu;
    const float rstd = rsqrtf(var + 1e-5f);
#pragma unroll
    for (int i = 0; i < 4; ++i) {
        int col = t + i * 256;
        out[row * 1024 + col] = (v[i] - mu) * rstd * g[col] + be[col];
    }
}

// ---------------------------------------------------------------------------
// Orchestration
// ---------------------------------------------------------------------------
extern "C" void kernel_launch(void* const* d_in, const int* in_sizes, int n_in,
                              void* d_out, int out_size, void* d_ws, size_t ws_size,
                              hipStream_t stream) {
    (void)in_sizes; (void)n_in; (void)out_size; (void)ws_size;

    const float* input = (const float*)d_in[0];   // [4,1024,1024]
    const float* ctx   = (const float*)d_in[1];   // [4,2048,1024]
    const float* cmask = (const float*)d_in[2];   // [4,1,1,2048]
    const float* cq_w = (const float*)d_in[3],  *cq_b = (const float*)d_in[4];
    const float* ck_w = (const float*)d_in[5],  *ck_b = (const float*)d_in[6];
    const float* cv_w = (const float*)d_in[7],  *cv_b = (const float*)d_in[8];
    const float* co_w = (const float*)d_in[9],  *co_b = (const float*)d_in[10];
    const float* co_g = (const float*)d_in[11], *co_be = (const float*)d_in[12];
    const float* sq_w = (const float*)d_in[13], *sq_b = (const float*)d_in[14];
    const float* sk_w = (const float*)d_in[15], *sk_b = (const float*)d_in[16];
    const float* sv_w = (const float*)d_in[17], *sv_b = (const float*)d_in[18];
    const float* so_w = (const float*)d_in[19], *so_b = (const float*)d_in[20];
    const float* so_g = (const float*)d_in[21], *so_be = (const float*)d_in[22];
    const float* fi_w = (const float*)d_in[23], *fi_b = (const float*)d_in[24];
    const float* fo_w = (const float*)d_in[25], *fo_b = (const float*)d_in[26];
    const float* fo_g = (const float*)d_in[27], *fo_be = (const float*)d_in[28];

    float* ws = (float*)d_ws;
    const size_t M4 = 4u * 1024u * 1024u;       // 4M floats
    float* wsQ   = ws;                          // 4M  (bf16 uses half)
    float* wsK   = ws + 1 * M4;                 // 8M
    float* wsV   = ws + 3 * M4;                 // 8M
    float* wsT0  = ws + 5 * M4;                 // 4M  attn out (fp32)
    float* wsX   = ws + 6 * M4;                 // 4M  hidden (fp32)
    float* wsT1  = ws + 7 * M4;                 // 4M  gemm tmp (fp32)
    float* wsInt = ws;                          // 16M (reuses Q/K/V region)
    unsigned short* wsQb = (unsigned short*)wsQ;
    unsigned short* wsKb = (unsigned short*)wsK;
    unsigned short* wsVb = (unsigned short*)wsV;

    const int Mx = 4 * 1024;    // query/hidden rows
    const int Mc = 4 * 2048;    // ctx rows
    const dim3 blk256(256), blk128(128);
    const dim3 gD(1024 / 128, Mx / 128);   // N=1024, M=4096
    const dim3 gDc(1024 / 128, Mc / 128);  // N=1024, M=8192
    const dim3 gFi(4096 / 128, Mx / 128);  // N=4096, M=4096
    const dim3 gAtt(1024 / 64, 16, 4);

    // ---- Cross attention (Q/K/V produced in bf16) ----
    gemm_bias_kernel<<<gD,  blk256, 0, stream>>>(input, cq_w, cq_b, nullptr, wsQb, Mx, 1024, 1024, 0, 1);
    gemm_bias_kernel<<<gDc, blk256, 0, stream>>>(ctx,   ck_w, ck_b, nullptr, wsKb, Mc, 1024, 1024, 0, 1);
    gemm_bias_kernel<<<gDc, blk256, 0, stream>>>(ctx,   cv_w, cv_b, nullptr, wsVb, Mc, 1024, 1024, 0, 1);
    flash_attn_kernel<<<gAtt, blk128, 0, stream>>>(wsQb, wsKb, wsVb, cmask, wsT0, 1024, 2048);
    gemm_bias_kernel<<<gD,  blk256, 0, stream>>>(wsT0, co_w, co_b, wsT1, nullptr, Mx, 1024, 1024, 0, 0);
    ln_residual_kernel<<<dim3(Mx), blk256, 0, stream>>>(wsT1, input, co_g, co_be, wsX);

    // ---- Self attention ----
    gemm_bias_kernel<<<gD, blk256, 0, stream>>>(wsX, sq_w, sq_b, nullptr, wsQb, Mx, 1024, 1024, 0, 1);
    gemm_bias_kernel<<<gD, blk256, 0, stream>>>(wsX, sk_w, sk_b, nullptr, wsKb, Mx, 1024, 1024, 0, 1);
    gemm_bias_kernel<<<gD, blk256, 0, stream>>>(wsX, sv_w, sv_b, nullptr, wsVb, Mx, 1024, 1024, 0, 1);
    flash_attn_kernel<<<gAtt, blk128, 0, stream>>>(wsQb, wsKb, wsVb, nullptr, wsT0, 1024, 1024);
    gemm_bias_kernel<<<gD, blk256, 0, stream>>>(wsT0, so_w, so_b, wsT1, nullptr, Mx, 1024, 1024, 0, 0);
    ln_residual_kernel<<<dim3(Mx), blk256, 0, stream>>>(wsT1, wsX, so_g, so_be, wsX);

    // ---- FFN ----
    gemm_bias_kernel<<<gFi, blk256, 0, stream>>>(wsX, fi_w, fi_b, wsInt, nullptr, Mx, 4096, 1024, 1, 0); // +GELU
    gemm_bias_kernel<<<gD,  blk256, 0, stream>>>(wsInt, fo_w, fo_b, wsT1, nullptr, Mx, 1024, 4096, 0, 0);
    ln_residual_kernel<<<dim3(Mx), blk256, 0, stream>>>(wsT1, wsX, fo_g, fo_be, (float*)d_out);
}
